// BERT_BiLSTM_CRF_50448685858820
// MI455X (gfx1250) — matrix operand order repaired and checked by reading, not compiled, not run on hardware
//
#include <hip/hip_runtime.h>

// ---------------- model dims ----------------
constexpr int kB   = 8;
constexpr int kS   = 256;
constexpr int kH   = 768;
constexpr int kNH  = 12;
constexpr int kDH  = 64;
constexpr int kFF  = 3072;
constexpr int kLH  = 256;
constexpr int kNL  = 12;
constexpr int kLAB = 9;
constexpr int kTOK = kB * kS;          // 2048

// ---------------- WMMA types ----------------
typedef __bf16 v16bf __attribute__((ext_vector_type(16)));
typedef __bf16 v2bf  __attribute__((ext_vector_type(2)));
typedef float  v8f   __attribute__((ext_vector_type(8)));
typedef float  v2f   __attribute__((ext_vector_type(2)));

union FragBF { v16bf v; unsigned int u[8]; };

// explicit global-address-space pointers so staging lowers to global_load_*
typedef const __attribute__((address_space(1))) float GF;
typedef const __attribute__((address_space(1))) v2f   GV2;

#if defined(__has_builtin)
#if __has_builtin(__builtin_amdgcn_cvt_pk_bf16_f32)
#define HAVE_CVT_PK_BF16 1
#endif
#endif

__device__ __forceinline__ unsigned int pack_bf16x2(float lo, float hi) {
#ifdef HAVE_CVT_PK_BF16
  union { v2bf v; unsigned int u; } c;
  c.v = __builtin_amdgcn_cvt_pk_bf16_f32(lo, hi);
  return c.u;
#else
  // lowers to a single v_cvt_pk_bf16_f32 on gfx1250 (verified in disasm)
  union { v2bf v; unsigned int u; } c;
  c.v[0] = (__bf16)lo;
  c.v[1] = (__bf16)hi;
  return c.u;
#endif
}

__device__ __forceinline__ float gelu_exact(float x) {
  return 0.5f * x * (1.0f + erff(x * 0.70710678118654752f));
}

// =====================================================================
// Generic strided-batched GEMM:  C = act(alpha * A · op(B) + bias)
//   TRANSB=1: B stored [N x K] (weight form, multiply by B^T)
//   TRANSB=0: B stored [K x N] (e.g. att·V)
// Block: 256 threads (8 waves), tile 64x64, BK=32, bf16 WMMA, f32 accum.
// TRANSB is a template parameter -> branch-free K-loop with constant
// pointer strides; staging pointers are global-AS and loop-invariant.
// =====================================================================
template <int TRANSB>
__global__ __launch_bounds__(256) void gemm_bf16_wmma(
    const float* __restrict__ A, const float* __restrict__ Bm,
    const float* __restrict__ bias, float* __restrict__ C,
    int K, int lda, int ldb, int ldc,
    int nH, long long sAb, long long sAh, long long sBb, long long sBh,
    long long sCb, long long sCh, int act, float alpha) {

  const int z  = blockIdx.z;
  const int bb = z / nH;
  const int hh = z - bb * nH;
  A  += (size_t)bb * sAb + (size_t)hh * sAh;
  Bm += (size_t)bb * sBb + (size_t)hh * sBh;
  C  += (size_t)bb * sCb + (size_t)hh * sCh;

  const int tid  = threadIdx.x;
  const int lane = tid & 31;
  const int wave = tid >> 5;
  const int wm   = wave & 3;        // 4 row-waves  (16 rows each)
  const int wn   = wave >> 2;       // 2 col-waves  (32 cols each)
  const int hl   = lane >> 4;       // lane half
  const int l15  = lane & 15;

  const int m0 = blockIdx.y * 64;
  const int n0 = blockIdx.x * 64;

  __shared__ unsigned int sA[64][17];   // [row][kpair] packed bf16x2
  __shared__ unsigned int sB[64][17];   // [col(N)][kpair] packed bf16x2

  // ---- loop-invariant staging addresses ----
  // idx = tid + 256*t  =>  kp = tid&15 (const), row/col = (tid>>4) + 16*t
  const int kp = tid & 15;
  const int r0 = tid >> 4;

  GF* pA[4];
  GF* pB[4];
  unsigned int* spA[4];
  unsigned int* spB[4];
#pragma unroll
  for (int t = 0; t < 4; ++t) {
    const int rc = r0 + 16 * t;
    pA[t]  = (GF*)(A + (size_t)(m0 + rc) * lda + kp * 2);
    spA[t] = &sA[rc][kp];
    spB[t] = &sB[rc][kp];
    if (TRANSB) pB[t] = (GF*)(Bm + (size_t)(n0 + rc) * ldb + kp * 2);
    else        pB[t] = (GF*)(Bm + (size_t)(kp * 2) * ldb + (n0 + rc));
  }
  const int stepB = TRANSB ? 32 : 32 * ldb;

  v8f acc0 = {};
  v8f acc1 = {};

  for (int k0 = 0; k0 < K; k0 += 32) {
    // ---- batched staging loads (clause-friendly: all loads first) ----
    v2f ra[4];
    v2f rb[4];
#pragma unroll
    for (int t = 0; t < 4; ++t) ra[t] = *(GV2*)pA[t];
    if (TRANSB) {
#pragma unroll
      for (int t = 0; t < 4; ++t) rb[t] = *(GV2*)pB[t];
    } else {
#pragma unroll
      for (int t = 0; t < 4; ++t) { rb[t].x = pB[t][0]; rb[t].y = pB[t][ldb]; }
    }
    // ---- convert + store to LDS ----
#pragma unroll
    for (int t = 0; t < 4; ++t) *spA[t] = pack_bf16x2(ra[t].x, ra[t].y);
#pragma unroll
    for (int t = 0; t < 4; ++t) *spB[t] = pack_bf16x2(rb[t].x, rb[t].y);
    // ---- advance staging pointers (compile-time-shaped stride) ----
#pragma unroll
    for (int t = 0; t < 4; ++t) { pA[t] += 32; pB[t] += stepB; }
    __syncthreads();

    // ---- assemble fragments per ISA lane layouts ----
    FragBF fa, fb0, fb1;
    const int arow = wm * 16 + l15;
#pragma unroll
    for (int j = 0; j < 4; ++j) {
      fa.u[j]     = sA[arow][j + hl * 4];       // K = 2j   (+8 upper half)
      fa.u[j + 4] = sA[arow][8 + j + hl * 4];   // K = 16+2j(+8 upper half)
    }
    const int col0 = wn * 32 + l15;
    const int col1 = col0 + 16;
#pragma unroll
    for (int j = 0; j < 4; ++j) {
      fb0.u[j]     = sB[col0][j + hl * 4];      // K = 2j   (+8 upper half)
      fb0.u[j + 4] = sB[col0][8 + j + hl * 4];  // K = 16+2j(+8 upper half)
      fb1.u[j]     = sB[col1][j + hl * 4];
      fb1.u[j + 4] = sB[col1][8 + j + hl * 4];
    }

    acc0 = __builtin_amdgcn_wmma_f32_16x16x32_bf16(
        false, fa.v, false, fb0.v, (short)0, acc0, false, false);
    acc1 = __builtin_amdgcn_wmma_f32_16x16x32_bf16(
        false, fa.v, false, fb1.v, (short)0, acc1, false, false);
    __syncthreads();
  }

  // ---- epilogue: scale, bias, activation, store f32 ----
#pragma unroll
  for (int rr = 0; rr < 8; ++rr) {
    const int row = m0 + wm * 16 + hl * 8 + rr;
#pragma unroll
    for (int ns = 0; ns < 2; ++ns) {
      const int col = n0 + wn * 32 + ns * 16 + l15;
      float v = (ns ? acc1[rr] : acc0[rr]) * alpha;
      if (bias) v += bias[col];
      if (act == 1) v = gelu_exact(v);
      C[(size_t)row * ldc + col] = v;
    }
  }
}

// =====================================================================
// Embedding sum: word + pos + type[0]
// =====================================================================
__global__ __launch_bounds__(256) void embed_kernel(
    const int* __restrict__ ids, const float* __restrict__ wemb,
    const float* __restrict__ pemb, const float* __restrict__ temb,
    float* __restrict__ out) {
  const int tok = blockIdx.x;
  const int s   = tok & (kS - 1);
  const int id  = ids[tok];
  const size_t ob = (size_t)tok * kH;
  for (int j = threadIdx.x; j < kH; j += 256)
    out[ob + j] = wemb[(size_t)id * kH + j] + pemb[(size_t)s * kH + j] + temb[j];
}

// =====================================================================
// LayerNorm over H=768, optional residual: out = LN(src + res) * g + b
// =====================================================================
__global__ __launch_bounds__(256) void layernorm_kernel(
    const float* __restrict__ src, const float* __restrict__ res,
    const float* __restrict__ g, const float* __restrict__ b,
    float* __restrict__ out) {
  const int tok = blockIdx.x;
  const int tid = threadIdx.x;
  const size_t base = (size_t)tok * kH;
  __shared__ float red[256];

  float v[3];
#pragma unroll
  for (int i = 0; i < 3; ++i) {
    int j = tid + i * 256;
    v[i] = src[base + j] + (res ? res[base + j] : 0.0f);
  }
  float s = v[0] + v[1] + v[2];
  red[tid] = s; __syncthreads();
  for (int w = 128; w > 0; w >>= 1) { if (tid < w) red[tid] += red[tid + w]; __syncthreads(); }
  const float mu = red[0] * (1.0f / kH);
  __syncthreads();
  float vs = 0.0f;
#pragma unroll
  for (int i = 0; i < 3; ++i) { float d = v[i] - mu; vs += d * d; }
  red[tid] = vs; __syncthreads();
  for (int w = 128; w > 0; w >>= 1) { if (tid < w) red[tid] += red[tid + w]; __syncthreads(); }
  const float rs = rsqrtf(red[0] * (1.0f / kH) + 1e-12f);
#pragma unroll
  for (int i = 0; i < 3; ++i) {
    int j = tid + i * 256;
    out[base + j] = (v[i] - mu) * rs * g[j] + b[j];
  }
}

// =====================================================================
// Masked softmax over last dim (S=256); one block per score row.
// =====================================================================
__global__ __launch_bounds__(256) void softmax_mask_kernel(
    float* __restrict__ scores, const int* __restrict__ mask) {
  const int row = blockIdx.x;
  const int b   = row / (kNH * kS);
  const int tid = threadIdx.x;
  float* p = scores + (size_t)row * kS;
  __shared__ float red[256];

  float v = p[tid] + (1.0f - (float)mask[b * kS + tid]) * -1e9f;
  red[tid] = v; __syncthreads();
  for (int w = 128; w > 0; w >>= 1) { if (tid < w) red[tid] = fmaxf(red[tid], red[tid + w]); __syncthreads(); }
  const float mx = red[0]; __syncthreads();
  float e = expf(v - mx);
  red[tid] = e; __syncthreads();
  for (int w = 128; w > 0; w >>= 1) { if (tid < w) red[tid] += red[tid + w]; __syncthreads(); }
  p[tid] = e / red[0];
}

// =====================================================================
// vector add (combined LSTM biases)
// =====================================================================
__global__ void add2_kernel(const float* a, const float* b, float* o, int n) {
  int i = blockIdx.x * blockDim.x + threadIdx.x;
  if (i < n) o[i] = a[i] + b[i];
}

// =====================================================================
// LSTM scan. One block per batch element.
// =====================================================================
__global__ __launch_bounds__(256) void lstm_scan_kernel(
    const float* __restrict__ xg, const float* __restrict__ whh,
    float* __restrict__ hs, int reverse) {
  const int b   = blockIdx.x;
  const int tid = threadIdx.x;
  __shared__ float sh[kLH], sc[kLH], sg[4 * kLH];
  sh[tid] = 0.0f; sc[tid] = 0.0f;
  __syncthreads();

  for (int step = 0; step < kS; ++step) {
    const int t = reverse ? (kS - 1 - step) : step;
    const float* xt = xg + (size_t)(b * kS + t) * (4 * kLH);
    for (int o = tid; o < 4 * kLH; o += 256) {
      float acc = xt[o];
      const float* w = whh + (size_t)o * kLH;
      for (int j = 0; j < kLH; ++j) acc += w[j] * sh[j];
      sg[o] = acc;
    }
    __syncthreads();
    {
      const float ig = 1.0f / (1.0f + expf(-sg[tid]));
      const float fg = 1.0f / (1.0f + expf(-sg[kLH + tid]));
      const float gg = tanhf(sg[2 * kLH + tid]);
      const float og = 1.0f / (1.0f + expf(-sg[3 * kLH + tid]));
      const float c  = fg * sc[tid] + ig * gg;
      const float h  = og * tanhf(c);
      sc[tid] = c; sh[tid] = h;
      hs[(size_t)(b * kS + t) * kLH + tid] = h;
    }
    __syncthreads();
  }
}

// =====================================================================
// fc: em[tok][lab] = [hf|hb] . fc_w[lab]^T + fc_b[lab]
// =====================================================================
__global__ void fc_kernel(const float* __restrict__ hf, const float* __restrict__ hb,
                          const float* __restrict__ w, const float* __restrict__ bias,
                          float* __restrict__ em) {
  int idx = blockIdx.x * blockDim.x + threadIdx.x;
  if (idx >= kTOK * kLAB) return;
  int tok = idx / kLAB, lab = idx - tok * kLAB;
  const float* wl = w + (size_t)lab * (2 * kLH);
  float acc = bias[lab];
  const float* pf = hf + (size_t)tok * kLH;
  const float* pb = hb + (size_t)tok * kLH;
  for (int j = 0; j < kLH; ++j) acc += pf[j] * wl[j];
  for (int j = 0; j < kLH; ++j) acc += pb[j] * wl[kLH + j];
  em[idx] = acc;
}

// =====================================================================
// CRF NLL per batch element (pytorch-crf semantics).
// =====================================================================
__global__ void crf_kernel(const float* __restrict__ em, const int* __restrict__ labels,
                           const int* __restrict__ mask, const float* __restrict__ trans,
                           const float* __restrict__ start, const float* __restrict__ endv,
                           float* __restrict__ nll) {
  const int b   = blockIdx.x;
  const int tid = threadIdx.x;
  const float* e = em + (size_t)b * kS * kLAB;
  const int* tg  = labels + b * kS;
  const int* mk  = mask + b * kS;
  __shared__ float alpha[kLAB], nxt[kLAB];

  if (tid < kLAB) alpha[tid] = start[tid] + e[tid];
  __syncthreads();
  for (int t = 1; t < kS; ++t) {
    if (tid < kLAB) {
      float mx = -3.0e38f;
      for (int i = 0; i < kLAB; ++i) mx = fmaxf(mx, alpha[i] + trans[i * kLAB + tid]);
      float s = 0.0f;
      for (int i = 0; i < kLAB; ++i) s += expf(alpha[i] + trans[i * kLAB + tid] - mx);
      nxt[tid] = mx + logf(s) + e[t * kLAB + tid];
    }
    __syncthreads();
    if (tid < kLAB && mk[t] > 0) alpha[tid] = nxt[tid];
    __syncthreads();
  }
  if (tid == 0) {
    float mx = -3.0e38f;
    for (int j = 0; j < kLAB; ++j) mx = fmaxf(mx, alpha[j] + endv[j]);
    float s = 0.0f;
    for (int j = 0; j < kLAB; ++j) s += expf(alpha[j] + endv[j] - mx);
    const float logZ = mx + logf(s);

    float sc = start[tg[0]] + e[tg[0]];
    int cnt = (mk[0] > 0) ? 1 : 0;
    for (int t = 1; t < kS; ++t) {
      if (mk[t] > 0) { sc += trans[tg[t - 1] * kLAB + tg[t]] + e[t * kLAB + tg[t]]; ++cnt; }
    }
    sc += endv[tg[cnt - 1]];
    nll[b] = logZ - sc;
  }
}

__global__ void mean_kernel(const float* __restrict__ nll, float* __restrict__ out) {
  if (threadIdx.x == 0) {
    float s = 0.0f;
    for (int i = 0; i < kB; ++i) s += nll[i];
    out[0] = s / (float)kB;
  }
}

// =====================================================================
// Host launcher
// =====================================================================
extern "C" void kernel_launch(void* const* d_in, const int* in_sizes, int n_in,
                              void* d_out, int out_size, void* d_ws, size_t ws_size,
                              hipStream_t stream) {
  (void)in_sizes; (void)n_in; (void)out_size; (void)ws_size;

  // ---- pytree flatten order (JAX dicts flatten in sorted-key order) ----
  auto F = [&](int i) -> const float* { return (const float*)d_in[i]; };
  const int* ids    = (const int*)d_in[0];
  const int* amask  = (const int*)d_in[1];
  const int* labels = (const int*)d_in[2];
  const float *crf_end = F(3), *crf_start = F(4), *crf_trans = F(5);
  const float *eln_b = F(6), *eln_g = F(7), *fc_b = F(8), *fc_w = F(9);
  const int LBASE = 10;
  const float *lb_bhh = F(202), *lb_bih = F(203), *lb_whh = F(204), *lb_wih = F(205);
  const float *lf_bhh = F(206), *lf_bih = F(207), *lf_whh = F(208), *lf_wih = F(209);
  const float *pemb = F(210), *temb = F(211), *wemb = F(212);

  // ---- workspace carve (floats) ----
  float* W = (float*)d_ws;
  size_t o = 0;
  float* x      = W + o; o += (size_t)kTOK * kH;
  float* t0     = W + o; o += (size_t)kTOK * kH;
  float* q      = W + o; o += (size_t)kTOK * kH;
  float* k      = W + o; o += (size_t)kTOK * kH;
  float* v      = W + o; o += (size_t)kTOK * kH;
  float* scores = W + o; o += (size_t)kB * kNH * kS * kS;
  float* ctx    = W + o; o += (size_t)kTOK * kH;
  float* ffh    = W + o; o += (size_t)kTOK * kFF;
  float* xg     = W + o; o += (size_t)kTOK * 4 * kLH;
  float* hf     = W + o; o += (size_t)kTOK * kLH;
  float* hb     = W + o; o += (size_t)kTOK * kLH;
  float* em     = W + o; o += (size_t)kTOK * kLAB;
  float* bias2  = W + o; o += 4 * kLH;
  float* nll    = W + o; o += kB;

  auto gemm = [&](const float* A, const float* Bm, const float* bias, float* C,
                  int M, int N, int K, int lda, int ldb, int ldc,
                  int batches, int nH,
                  long long sAb, long long sAh, long long sBb, long long sBh,
                  long long sCb, long long sCh, int transB, int act, float alpha) {
    dim3 g(N / 64, M / 64, batches);
    if (transB)
      gemm_bf16_wmma<1><<<g, dim3(256), 0, stream>>>(
          A, Bm, bias, C, K, lda, ldb, ldc, nH,
          sAb, sAh, sBb, sBh, sCb, sCh, act, alpha);
    else
      gemm_bf16_wmma<0><<<g, dim3(256), 0, stream>>>(
          A, Bm, bias, C, K, lda, ldb, ldc, nH,
          sAb, sAh, sBb, sBh, sCb, sCh, act, alpha);
  };

  const long long sTok = (long long)kS * kH;
  const long long sSc  = (long long)kS * kS;

  // ---- embeddings + LN ----
  embed_kernel<<<kTOK, 256, 0, stream>>>(ids, wemb, pemb, temb, t0);
  layernorm_kernel<<<kTOK, 256, 0, stream>>>(t0, nullptr, eln_g, eln_b, x);

  // ---- encoder layers ----
  for (int l = 0; l < kNL; ++l) {
    const int P = LBASE + 16 * l;
    const float *ff1_b=F(P+0), *ff1_w=F(P+1), *ff2_b=F(P+2), *ff2_w=F(P+3);
    const float *k_b=F(P+4), *k_w=F(P+5), *ln1_b=F(P+6), *ln1_g=F(P+7);
    const float *ln2_b=F(P+8), *ln2_g=F(P+9), *o_b=F(P+10), *o_w=F(P+11);
    const float *q_b=F(P+12), *q_w=F(P+13), *v_b=F(P+14), *v_w=F(P+15);

    gemm(x, q_w, q_b, q, kTOK, kH, kH, kH, kH, kH, 1, 1, 0,0,0,0,0,0, 1, 0, 1.0f);
    gemm(x, k_w, k_b, k, kTOK, kH, kH, kH, kH, kH, 1, 1, 0,0,0,0,0,0, 1, 0, 1.0f);
    gemm(x, v_w, v_b, v, kTOK, kH, kH, kH, kH, kH, 1, 1, 0,0,0,0,0,0, 1, 0, 1.0f);

    gemm(q, k, nullptr, scores, kS, kS, kDH, kH, kH, kS,
         kB * kNH, kNH, sTok, kDH, sTok, kDH, (long long)kNH * sSc, sSc,
         1, 0, 0.125f);
    softmax_mask_kernel<<<kB * kNH * kS, 256, 0, stream>>>(scores, amask);
    gemm(scores, v, nullptr, ctx, kS, kDH, kS, kS, kH, kH,
         kB * kNH, kNH, (long long)kNH * sSc, sSc, sTok, kDH, sTok, kDH,
         0, 0, 1.0f);

    gemm(ctx, o_w, o_b, t0, kTOK, kH, kH, kH, kH, kH, 1, 1, 0,0,0,0,0,0, 1, 0, 1.0f);
    layernorm_kernel<<<kTOK, 256, 0, stream>>>(t0, x, ln1_g, ln1_b, x);

    gemm(x, ff1_w, ff1_b, ffh, kTOK, kFF, kH, kH, kH, kFF, 1, 1, 0,0,0,0,0,0, 1, 1, 1.0f);
    gemm(ffh, ff2_w, ff2_b, t0, kTOK, kH, kFF, kFF, kFF, kH, 1, 1, 0,0,0,0,0,0, 1, 0, 1.0f);
    layernorm_kernel<<<kTOK, 256, 0, stream>>>(t0, x, ln2_g, ln2_b, x);
  }

  // ---- BiLSTM ----
  add2_kernel<<<4, 256, 0, stream>>>(lf_bih, lf_bhh, bias2, 4 * kLH);
  gemm(x, lf_wih, bias2, xg, kTOK, 4 * kLH, kH, kH, kH, 4 * kLH,
       1, 1, 0,0,0,0,0,0, 1, 0, 1.0f);
  lstm_scan_kernel<<<kB, 256, 0, stream>>>(xg, lf_whh, hf, 0);

  add2_kernel<<<4, 256, 0, stream>>>(lb_bih, lb_bhh, bias2, 4 * kLH);
  gemm(x, lb_wih, bias2, xg, kTOK, 4 * kLH, kH, kH, kH, 4 * kLH,
       1, 1, 0,0,0,0,0,0, 1, 0, 1.0f);
  lstm_scan_kernel<<<kB, 256, 0, stream>>>(xg, lb_whh, hb, 1);

  // ---- emissions + CRF ----
  fc_kernel<<<(kTOK * kLAB + 255) / 256, 256, 0, stream>>>(hf, hb, fc_w, fc_b, em);
  crf_kernel<<<kB, 32, 0, stream>>>(em, labels, amask, crf_trans, crf_start, crf_end, nll);
  mean_kernel<<<1, 32, 0, stream>>>(nll, (float*)d_out);
}
